// MultiHeadAttention_67190468378948
// MI455X (gfx1250) — matrix operand order repaired and checked by reading, not compiled, not run on hardware
//
#include <hip/hip_runtime.h>
#include <hip/hip_bf16.h>
#include <math.h>

#define B_  4
#define S_  2048
#define D_  1024
#define H_  16
#define HD_ 64

typedef __attribute__((ext_vector_type(16))) _Float16 v16h;
typedef __attribute__((ext_vector_type(8)))  _Float16 v8h;
typedef __attribute__((ext_vector_type(4)))  _Float16 v4h;
typedef __attribute__((ext_vector_type(8)))  float    v8f;
typedef __attribute__((ext_vector_type(4)))  float    v4f;

static __device__ __forceinline__ v16h concat8(v8h lo, v8h hi) {
  v16h r;
#pragma unroll
  for (int i = 0; i < 8; ++i) { r[i] = lo[i]; r[i + 8] = hi[i]; }
  return r;
}

static __device__ __forceinline__ v8f wmma_f16(v16h a, v16h b, v8f c) {
  // (neg_a, A, neg_b, B, c_mod, C, reuse_a, reuse_b)
  return __builtin_amdgcn_wmma_f32_16x16x32_f16(false, a, false, b, (short)0, c,
                                                false, false);
}

// ---------------------------------------------------------------------------
// Kernel 1: fused QKV projection.  out = x @ W + b (f16).
// Q,K scattered to [B,H,S,HD]; V scattered TRANSPOSED to [B,H,HD,S].
// Block: 256 threads = 8 waves; tile 128(M) x 128(N), K-step 32.
// Waves 4(M) x 2(N); each wave owns 32x64 = 2x4 WMMA accumulators.
// Staging is register-double-buffered: the next K-tile's global loads are
// issued right after the barrier so they are in flight under the 8 WMMAs.
// ---------------------------------------------------------------------------
__global__ __launch_bounds__(256) void qkv_gemm_kernel(
    const float* __restrict__ x,
    const float* __restrict__ Wq, const float* __restrict__ bq,
    const float* __restrict__ Wk, const float* __restrict__ bk,
    const float* __restrict__ Wv, const float* __restrict__ bv,
    _Float16* __restrict__ Qh, _Float16* __restrict__ Kh,
    _Float16* __restrict__ Vt) {
  __shared__ _Float16 ldsA[128 * 40];  // row-major [m][k], stride 40 halves
  __shared__ _Float16 ldsB[128 * 40];  // col-major [n][k], stride 40 halves

  const int t = threadIdx.x;
  const int wave = t >> 5, lane = t & 31;
  const int lhalf = lane >> 4;  // 0: lanes 0-15, 1: lanes 16-31
  const int l16 = lane & 15;
  const int waveM = wave >> 1, waveN = wave & 1;

  const int rowbase = blockIdx.y * 128;  // over B*S
  const int nbase = blockIdx.x * 128;    // over D
  const int z = blockIdx.z;
  const float* W = (z == 0) ? Wq : (z == 1) ? Wk : Wv;
  const float* bias = (z == 0) ? bq : (z == 1) ? bk : bv;
  _Float16* dst = (z == 0) ? Qh : (z == 1) ? Kh : Vt;

  // Per-thread staging coordinates (fixed across K-steps).
  const int a_r = (t * 4) >> 5, a_c = (t * 4) & 31;       // + i*32 rows
  const int b_k = (t * 4) >> 7, b_n = (t * 4) & 127;      // + i*8 k-rows

  v8f acc[2][4];
#pragma unroll
  for (int i = 0; i < 2; ++i)
#pragma unroll
    for (int j = 0; j < 4; ++j) acc[i][j] = (v8f)0.0f;

  // Preload first K-tile into registers (8 clustered b128 loads).
  v4f va[4], vb[4];
#pragma unroll
  for (int i = 0; i < 4; ++i)
    va[i] = *(const v4f*)&x[(size_t)(rowbase + a_r + i * 32) * D_ + a_c];
#pragma unroll
  for (int i = 0; i < 4; ++i)
    vb[i] = *(const v4f*)&W[(size_t)(b_k + i * 8) * D_ + nbase + b_n];

  for (int kk = 0; kk < D_; kk += 32) {
    // Convert staged registers -> LDS.
#pragma unroll
    for (int i = 0; i < 4; ++i) {
      v4h hv;
#pragma unroll
      for (int j = 0; j < 4; ++j) hv[j] = (_Float16)va[i][j];
      *(v4h*)&ldsA[(a_r + i * 32) * 40 + a_c] = hv;
    }
#pragma unroll
    for (int i = 0; i < 4; ++i) {
#pragma unroll
      for (int j = 0; j < 4; ++j)
        ldsB[(b_n + j) * 40 + b_k + i * 8] = (_Float16)vb[i][j];
    }
    __syncthreads();

    // Issue next K-tile's loads now; they complete under the WMMAs.
    if (kk + 32 < D_) {
#pragma unroll
      for (int i = 0; i < 4; ++i)
        va[i] = *(const v4f*)&x[(size_t)(rowbase + a_r + i * 32) * D_ + kk +
                                32 + a_c];
#pragma unroll
      for (int i = 0; i < 4; ++i)
        vb[i] = *(const v4f*)&W[(size_t)(kk + 32 + b_k + i * 8) * D_ + nbase +
                                b_n];
    }

    // A fragment (ISA 16-bit A 16x32 layout): lane row M=l16; halves 0-7 ->
    // K=lhalf*8+0..7, halves 8-15 -> K=16+lhalf*8+0..7.
    v16h aF[2], bF[4];
#pragma unroll
    for (int mt = 0; mt < 2; ++mt) {
      int row = waveM * 32 + mt * 16 + l16;
      int o = lhalf * 8;
      aF[mt] = concat8(*(const v8h*)&ldsA[row * 40 + o],
                       *(const v8h*)&ldsA[row * 40 + 16 + o]);
    }
    // B fragment: lane col N=l16; halves 0-7 -> K=lhalf*8+0..7, halves 8-15
    // -> K=16+lhalf*8+0..7 (LDS B tile is column-major).
#pragma unroll
    for (int nt = 0; nt < 4; ++nt) {
      int col = waveN * 64 + nt * 16 + l16;
      int o = lhalf * 8;
      bF[nt] = concat8(*(const v8h*)&ldsB[col * 40 + o],
                       *(const v8h*)&ldsB[col * 40 + o + 16]);
    }
#pragma unroll
    for (int mt = 0; mt < 2; ++mt)
#pragma unroll
      for (int nt = 0; nt < 4; ++nt)
        acc[mt][nt] = wmma_f16(aF[mt], bF[nt], acc[mt][nt]);
    __syncthreads();
  }

  // Epilogue: C layout VGPR r -> M=r+lhalf*8, N=l16.  Bias, f16 scatter.
#pragma unroll
  for (int mt = 0; mt < 2; ++mt) {
#pragma unroll
    for (int nt = 0; nt < 4; ++nt) {
      int n = nbase + waveN * 64 + nt * 16 + l16;
      float bb = bias[n];
      int h = n >> 6, hd = n & 63;
#pragma unroll
      for (int r = 0; r < 8; ++r) {
        int m = rowbase + waveM * 32 + mt * 16 + r + lhalf * 8;
        int b = m >> 11;  // / S_
        int s = m & (S_ - 1);
        _Float16 val = (_Float16)(acc[mt][nt][r] + bb);
        if (z == 2)  // V transposed: [B,H,HD,S]
          dst[(((size_t)b * H_ + h) * HD_ + hd) * S_ + s] = val;
        else         // Q,K: [B,H,S,HD]
          dst[(((size_t)b * H_ + h) * S_ + s) * HD_ + hd] = val;
      }
    }
  }
}

// ---------------------------------------------------------------------------
// Kernel 2: causal flash attention.  grid = (S/128, B*H), 8 waves/block,
// each wave owns 16 query rows.  QK^T and PV on WMMA; V is pre-transposed so
// both score and PV B-fragments are contiguous 16-byte global loads.
// ---------------------------------------------------------------------------
__global__ __launch_bounds__(256) void attn_kernel(
    const _Float16* __restrict__ Qh, const _Float16* __restrict__ Kh,
    const _Float16* __restrict__ Vt, _Float16* __restrict__ Ch) {
  __shared__ _Float16 ldsP[8 * 16 * 40];  // per-wave private 16x32 P tile

  const int t = threadIdx.x;
  const int wave = t >> 5, lane = t & 31;
  const int lhalf = lane >> 4, l16 = lane & 15;
  const int bh = blockIdx.y;
  const int bidx = bh >> 4, h = bh & (H_ - 1);
  const int qbase = blockIdx.x * 128 + wave * 16;

  const _Float16* Qp = Qh + (size_t)bh * S_ * HD_;
  const _Float16* Kp = Kh + (size_t)bh * S_ * HD_;
  const _Float16* Vtp = Vt + (size_t)bh * HD_ * S_;  // [hd][s]
  _Float16* myP = &ldsP[wave * 16 * 40];

  // Q fragments (fixed for whole kernel): j=0 -> hd 0..31, j=1 -> 32..63
  v16h aQ[2];
  {
    const _Float16* qr = Qp + (size_t)(qbase + l16) * HD_;
#pragma unroll
    for (int j = 0; j < 2; ++j) {
      int o = j * 32 + lhalf * 8;
      aQ[j] = concat8(*(const v8h*)&qr[o], *(const v8h*)&qr[o + 16]);
    }
  }

  float m_run[8], l_run[8];
#pragma unroll
  for (int r = 0; r < 8; ++r) {
    m_run[r] = -__builtin_inff();
    l_run[r] = 0.0f;
  }
  v8f acc[4];
#pragma unroll
  for (int c = 0; c < 4; ++c) acc[c] = (v8f)0.0f;

  const int nk = (qbase + 47) >> 5;  // cover keys 0 .. qbase+15 (causal)
  for (int kb = 0; kb < nk; ++kb) {
    const int kbase = kb * 32;

    if (kb + 1 < nk) {  // warm cache for next key block (K rows + V rows)
      __builtin_prefetch(Kp + (size_t)(kbase + 32 + lane) * HD_, 0, 1);
      __builtin_prefetch(Vtp + (size_t)(lane * 2) * S_ + kbase + 32, 0, 1);
      __builtin_prefetch(Vtp + (size_t)(lane * 2 + 1) * S_ + kbase + 32, 0, 1);
    }

    // ---- scores: two 16x16 tiles over 32 keys, K-dim = HD via 2 WMMAs ----
    v8f sc[2];
#pragma unroll
    for (int tt = 0; tt < 2; ++tt) {
      const _Float16* krow = Kp + (size_t)(kbase + tt * 16 + l16) * HD_;
      // B (K^T) fragment: N=key=l16, halves 0-7 -> hd = lhalf*8 + e, 8-15 -> 16 + lhalf*8 + e
      v16h b0 = concat8(*(const v8h*)&krow[lhalf * 8],
                        *(const v8h*)&krow[lhalf * 8 + 16]);
      v16h b1 = concat8(*(const v8h*)&krow[32 + lhalf * 8],
                        *(const v8h*)&krow[32 + lhalf * 8 + 16]);
      v8f s = (v8f)0.0f;
      s = wmma_f16(aQ[0], b0, s);
      s = wmma_f16(aQ[1], b1, s);
      sc[tt] = s;
    }

    // ---- scale by 1/sqrt(HD) and causal mask ----
#pragma unroll
    for (int tt = 0; tt < 2; ++tt) {
      int key = kbase + tt * 16 + l16;
#pragma unroll
      for (int r = 0; r < 8; ++r) {
        int row = qbase + r + lhalf * 8;
        float v = sc[tt][r] * 0.125f;
        sc[tt][r] = (key > row) ? -__builtin_inff() : v;
      }
    }

    // ---- online softmax (row stats replicated across each 16-lane half) ----
    float pv0[8], pv1[8];
#pragma unroll
    for (int r = 0; r < 8; ++r) {
      float bm = fmaxf(sc[0][r], sc[1][r]);
#pragma unroll
      for (int i = 1; i < 16; i <<= 1) bm = fmaxf(bm, __shfl_xor(bm, i, 16));
      float mn = fmaxf(m_run[r], bm);       // finite from kb=0 (key 0 unmasked)
      float scale = __expf(m_run[r] - mn);  // exp(-inf)=0 on first block
      float p0 = __expf(sc[0][r] - mn);
      float p1 = __expf(sc[1][r] - mn);
      float ps = p0 + p1;
#pragma unroll
      for (int i = 1; i < 16; i <<= 1) ps += __shfl_xor(ps, i, 16);
      l_run[r] = l_run[r] * scale + ps;
      m_run[r] = mn;
#pragma unroll
      for (int c = 0; c < 4; ++c) acc[c][r] = acc[c][r] * scale;
      pv0[r] = p0;
      pv1[r] = p1;
    }

    // ---- C-layout -> A-layout for P via per-wave-private LDS tile ----
    // (same-wave RAW ordered by DScnt; no cross-wave sharing, no barrier)
#pragma unroll
    for (int r = 0; r < 8; ++r) {
      int row = r + lhalf * 8;
      myP[row * 40 + l16] = (_Float16)pv0[r];
      myP[row * 40 + 16 + l16] = (_Float16)pv1[r];
    }
    v16h aP;
    {
      int o = lhalf * 8;
      aP = concat8(*(const v8h*)&myP[l16 * 40 + o],
                   *(const v8h*)&myP[l16 * 40 + 16 + o]);
    }

    // ---- PV: 4 WMMAs over hd chunks; V^T rows give contiguous B-frags ----
#pragma unroll
    for (int c = 0; c < 4; ++c) {
      const _Float16* vrow =
          Vtp + (size_t)(c * 16 + l16) * S_ + kbase + lhalf * 8;
      v16h bV = concat8(*(const v8h*)&vrow[0], *(const v8h*)&vrow[16]);
      acc[c] = wmma_f16(aP, bV, acc[c]);
    }
  }

  // ---- normalize and write ctx as f16 into [B, S, D] ----
#pragma unroll
  for (int c = 0; c < 4; ++c) {
#pragma unroll
    for (int r = 0; r < 8; ++r) {
      int row = qbase + r + lhalf * 8;  // s index
      int hd = c * 16 + l16;
      float v = acc[c][r] / l_run[r];
      Ch[((size_t)bidx * S_ + row) * D_ + h * HD_ + hd] = (_Float16)v;
    }
  }
}

// ---------------------------------------------------------------------------
// Kernel 3: output projection.  out = ctx(f16) @ Wo + bo, f32 out.
// Same 128x128 tiling and register-double-buffered staging as kernel 1;
// A is already f16 (pure v8h copies).
// ---------------------------------------------------------------------------
__global__ __launch_bounds__(256) void oproj_kernel(
    const _Float16* __restrict__ Ch, const float* __restrict__ Wo,
    const float* __restrict__ bo, float* __restrict__ out) {
  __shared__ _Float16 ldsA[128 * 40];
  __shared__ _Float16 ldsB[128 * 40];

  const int t = threadIdx.x;
  const int wave = t >> 5, lane = t & 31;
  const int lhalf = lane >> 4, l16 = lane & 15;
  const int waveM = wave >> 1, waveN = wave & 1;
  const int rowbase = blockIdx.y * 128;
  const int nbase = blockIdx.x * 128;

  const int a_r = (t * 8) >> 5, a_c = (t * 8) & 31;   // + i*64 rows
  const int b_k = (t * 4) >> 7, b_n = (t * 4) & 127;  // + i*8 k-rows

  v8f acc[2][4];
#pragma unroll
  for (int i = 0; i < 2; ++i)
#pragma unroll
    for (int j = 0; j < 4; ++j) acc[i][j] = (v8f)0.0f;

  v8h va[2];
  v4f vb[4];
#pragma unroll
  for (int i = 0; i < 2; ++i)
    va[i] = *(const v8h*)&Ch[(size_t)(rowbase + a_r + i * 64) * D_ + a_c];
#pragma unroll
  for (int i = 0; i < 4; ++i)
    vb[i] = *(const v4f*)&Wo[(size_t)(b_k + i * 8) * D_ + nbase + b_n];

  for (int kk = 0; kk < D_; kk += 32) {
#pragma unroll
    for (int i = 0; i < 2; ++i)
      *(v8h*)&ldsA[(a_r + i * 64) * 40 + a_c] = va[i];
#pragma unroll
    for (int i = 0; i < 4; ++i) {
#pragma unroll
      for (int j = 0; j < 4; ++j)
        ldsB[(b_n + j) * 40 + b_k + i * 8] = (_Float16)vb[i][j];
    }
    __syncthreads();

    if (kk + 32 < D_) {
#pragma unroll
      for (int i = 0; i < 2; ++i)
        va[i] = *(const v8h*)&Ch[(size_t)(rowbase + a_r + i * 64) * D_ + kk +
                                 32 + a_c];
#pragma unroll
      for (int i = 0; i < 4; ++i)
        vb[i] = *(const v4f*)&Wo[(size_t)(kk + 32 + b_k + i * 8) * D_ + nbase +
                                 b_n];
    }

    v16h aF[2], bF[4];
#pragma unroll
    for (int mt = 0; mt < 2; ++mt) {
      int row = waveM * 32 + mt * 16 + l16;
      int o = lhalf * 8;
      aF[mt] = concat8(*(const v8h*)&ldsA[row * 40 + o],
                       *(const v8h*)&ldsA[row * 40 + 16 + o]);
    }
#pragma unroll
    for (int nt = 0; nt < 4; ++nt) {
      int col = waveN * 64 + nt * 16 + l16;
      int o = lhalf * 8;
      bF[nt] = concat8(*(const v8h*)&ldsB[col * 40 + o],
                       *(const v8h*)&ldsB[col * 40 + o + 16]);
    }
#pragma unroll
    for (int mt = 0; mt < 2; ++mt)
#pragma unroll
      for (int nt = 0; nt < 4; ++nt)
        acc[mt][nt] = wmma_f16(aF[mt], bF[nt], acc[mt][nt]);
    __syncthreads();
  }

#pragma unroll
  for (int mt = 0; mt < 2; ++mt) {
#pragma unroll
    for (int nt = 0; nt < 4; ++nt) {
      int n = nbase + waveN * 64 + nt * 16 + l16;
      float bb = bo[n];
#pragma unroll
      for (int r = 0; r < 8; ++r) {
        int m = rowbase + waveM * 32 + mt * 16 + r + lhalf * 8;
        out[(size_t)m * D_ + n] = acc[mt][nt][r] + bb;
      }
    }
  }
}

// ---------------------------------------------------------------------------
extern "C" void kernel_launch(void* const* d_in, const int* in_sizes, int n_in,
                              void* d_out, int out_size, void* d_ws,
                              size_t ws_size, hipStream_t stream) {
  const float* x = (const float*)d_in[0];
  const float* Wq = (const float*)d_in[1];
  const float* bq = (const float*)d_in[2];
  const float* Wk = (const float*)d_in[3];
  const float* bk = (const float*)d_in[4];
  const float* Wv = (const float*)d_in[5];
  const float* bv = (const float*)d_in[6];
  const float* Wo = (const float*)d_in[7];
  const float* bo = (const float*)d_in[8];
  float* out = (float*)d_out;

  const size_t perTensor = (size_t)B_ * S_ * D_;  // 8M halves each
  _Float16* Qh = (_Float16*)d_ws;
  _Float16* Kh = Qh + perTensor;
  _Float16* Vt = Kh + perTensor;  // [B,H,HD,S]
  _Float16* Ch = Vt + perTensor;  // 64 MB total workspace

  dim3 blk(256);
  dim3 g1(D_ / 128, (B_ * S_) / 128, 3);
  qkv_gemm_kernel<<<g1, blk, 0, stream>>>(x, Wq, bq, Wk, bk, Wv, bv, Qh, Kh,
                                          Vt);
  dim3 g2(S_ / 128, B_ * H_);
  attn_kernel<<<g2, blk, 0, stream>>>(Qh, Kh, Vt, Ch);
  dim3 g3(D_ / 128, (B_ * S_) / 128);
  oproj_kernel<<<g3, blk, 0, stream>>>(Ch, Wo, bo, out);
}
